// UnifiedASAAttention_3659312136324
// MI455X (gfx1250) — matrix operand order repaired and checked by reading, not compiled, not run on hardware
//
#include <hip/hip_runtime.h>

// ---------------------------------------------------------------------------
// UnifiedASA attention for MI455X (gfx1250, wave32, WMMA bf16 + TDM staging)
// B=64, N=512, D=1024, SHELL=64, RFF=32, ORDER=4, COMBINED=64
// ---------------------------------------------------------------------------

typedef __attribute__((ext_vector_type(16))) __bf16 v16bf;
typedef __attribute__((ext_vector_type(8)))  __bf16 v8bf;
typedef __attribute__((ext_vector_type(8)))  float  v8f;
typedef __attribute__((ext_vector_type(4)))  unsigned int v4u;
typedef __attribute__((ext_vector_type(8)))  int  v8i;
typedef __attribute__((ext_vector_type(4)))  int  v4i;

static constexpr int kB     = 64;
static constexpr int kN     = 512;
static constexpr int kD     = 1024;
static constexpr int kShell = 64;
static constexpr int kRFF   = 32;

#if defined(__AMDGCN__) && __has_builtin(__builtin_amdgcn_tensor_load_to_lds) && \
    __has_builtin(__builtin_amdgcn_s_wait_tensorcnt)
#define ASA_USE_TDM 1
#else
#define ASA_USE_TDM 0
#endif

__device__ __forceinline__ __bf16 to_bf16(float f) {
  union { float f; unsigned u; } v; v.f = f;
  unsigned r = (v.u + 0x7FFFu + ((v.u >> 16) & 1u)) >> 16;
  union { unsigned short s; __bf16 b; } o; o.s = (unsigned short)r;
  return o.b;
}

__device__ __forceinline__ float wave_sum32(float v) {
#pragma unroll
  for (int o = 16; o > 0; o >>= 1) v += __shfl_xor(v, o, 32);
  return v;
}
__device__ __forceinline__ float half_max16(float v) {
#pragma unroll
  for (int o = 8; o > 0; o >>= 1) v = fmaxf(v, __shfl_xor(v, o, 32));
  return v;
}
__device__ __forceinline__ float half_sum16(float v) {
#pragma unroll
  for (int o = 8; o > 0; o >>= 1) v += __shfl_xor(v, o, 32);
  return v;
}

// --- WMMA fragment loaders (contiguous k order k = 16*half + s, same for A and B, 16-bit, wave32) ---
// A: 16x32 (MxK) row-major.  lane<16: K in {0..15}; lane>=16: {16..31}
__device__ __forceinline__ v16bf load_a_frag(const __bf16* __restrict__ A,
                                             int lda, int kbase) {
  const int lane = threadIdx.x & 31;
  const int half = lane >> 4;
  const int m    = lane & 15;
  const __bf16* r = A + (long long)m * lda + kbase + half * 16;
  v8bf lo = *(const v8bf*)(r);        // K = kbase + half*16 + 0..7
  v8bf hi = *(const v8bf*)(r + 8);    // K = kbase + half*16 + 8..15
  return __builtin_shufflevector(lo, hi, 0,1,2,3,4,5,6,7,8,9,10,11,12,13,14,15);
}
// B: 32x16 (KxN) operand taken from an (N,K) row-major matrix (computes A*B^T).
__device__ __forceinline__ v16bf load_b_frag_nt(const __bf16* __restrict__ Bw,
                                                int ldb, int kbase) {
  const int lane  = threadIdx.x & 31;
  const int n     = lane & 15;
  const int khalf = lane >> 4;
  const __bf16* r = Bw + (long long)n * ldb + kbase + khalf * 16;
  return *(const v16bf*)(r);
}
// Same fragment, but from an LDS-staged 64x32 tile (row = n-local, col = k-local)
__device__ __forceinline__ v16bf load_b_frag_lds(const __bf16* Bsh, int t) {
  const int lane  = threadIdx.x & 31;
  const int n     = lane & 15;
  const int khalf = lane >> 4;
  return *(const v16bf*)(Bsh + (t * 16 + n) * 32 + khalf * 16);
}

__device__ __forceinline__ v8f wmma_bf16(v16bf a, v16bf b, v8f c) {
  return __builtin_amdgcn_wmma_f32_16x16x32_bf16(false, a, false, b,
                                                 (short)0, c, false, false);
}

#if ASA_USE_TDM
// Issue a TDM load of a 64-row x 32-elem bf16 tile (row stride K elems) to LDS.
// D# bitfields per CDNA5 ISA 8.3/8.4 (group0: count/lds/global/type,
// group1: data_size, tensor dims, tile dims, dim0 stride).
__device__ __forceinline__ void tdm_load_b_tile(const __bf16* gsrc,
                                                unsigned ldsAddr, int K) {
  unsigned long long ga = (unsigned long long)gsrc;
  v4u g0;
  g0.x = 1u;                                        // count=1 (valid descriptor)
  g0.y = ldsAddr;                                   // lds_addr (bytes)
  g0.z = (unsigned)(ga & 0xFFFFFFFFull);            // global_addr[31:0]
  g0.w = (unsigned)((ga >> 32) & 0x01FFFFFFull)     // global_addr[56:32]
         | 0x80000000u;                             // type=2 ("image")
  v8i g1;
  g1[0] = 0x00010000;                 // workgroup_mask=0, data_size=1 (2 bytes)
  g1[1] = (K & 0xFFFF) << 16;         // tensor_dim0 = K (lo16)
  g1[2] = (int)0xFFFF0000;            // tensor_dim0 hi=0, tensor_dim1 = 0xFFFF
  g1[3] = 32 << 16;                   // tensor_dim1 hi=0, tile_dim0 = 32
  g1[4] = 64;                         // tile_dim1 = 64, tile_dim2 = 0
  g1[5] = K;                          // tensor_dim0_stride lo32 = K
  g1[6] = 0;                          // stride0 hi16 = 0, stride1 lo16 = 0
  g1[7] = 0;
  v4i gz4 = {0, 0, 0, 0};
  v8i gz8 = {0, 0, 0, 0, 0, 0, 0, 0};
  // 6-arg toolchain form: (group0, group1, group2, group3, extra, cpol)
  __builtin_amdgcn_tensor_load_to_lds(g0, g1, gz4, gz4, gz8, 0);
}
#endif

// Fallback cooperative copy: 256 threads move the 64x32 bf16 tile (4 KB).
__device__ __forceinline__ void coop_load_b_tile(const __bf16* __restrict__ gsrc,
                                                 __bf16* lds, int K) {
  const int t = threadIdx.x;
  const int r = t >> 2;
  const int c = (t & 3) * 8;
  *(v8bf*)(lds + r * 32 + c) = *(const v8bf*)(gsrc + (long long)r * K + c);
}

// ---------------------------------------------------------------------------
// f32 -> bf16 conversion
// ---------------------------------------------------------------------------
__global__ __launch_bounds__(256) void cvt_f32_bf16_kernel(
    const float* __restrict__ in, __bf16* __restrict__ out, long long n) {
  long long i = (long long)blockIdx.x * 256 + threadIdx.x;
  if (i < n) out[i] = to_bf16(in[i]);
}

// ---------------------------------------------------------------------------
// Per-token feature pipeline -> phi (B*N, 64) bf16.  One wave per token.
// ---------------------------------------------------------------------------
__global__ __launch_bounds__(256) void features_kernel(
    const float* __restrict__ x,
    const float* __restrict__ Wc,  const float* __restrict__ bc,
    const float* __restrict__ Ws,  const float* __restrict__ bs,
    const float* __restrict__ omega, const float* __restrict__ rbias,
    const float* __restrict__ Wcomb,
    __bf16* __restrict__ phiOut) {
  const int wave = threadIdx.x >> 5;
  const int lane = threadIdx.x & 31;
  const int tok  = blockIdx.x * 8 + wave;
  const float* xr = x + (long long)tok * kD;

  float xv[32];
#pragma unroll
  for (int i = 0; i < 32; ++i) xv[i] = xr[lane + 32 * i];

  float qa = 0.f;
#pragma unroll
  for (int i = 0; i < 32; ++i) qa = fmaf(xv[i], Wc[lane + 32 * i], qa);
  qa = wave_sum32(qa) + bc[0];
  const float q = tanhf(qa);

  float sh0 = 0.f, sh1 = 0.f, nrm = 0.f;
  for (int j = 0; j < kShell; ++j) {
    const float* wr = Ws + (long long)j * kD;
    float s = 0.f;
#pragma unroll
    for (int i = 0; i < 32; ++i) s = fmaf(xv[i], wr[lane + 32 * i], s);
    s = wave_sum32(s) + bs[j];
    nrm = fmaf(s, s, nrm);
    if ((j >> 1) == lane) { if (j & 1) sh1 = s; else sh0 = s; }
  }
  const float inv = 1.0f / fmaxf(sqrtf(nrm), 1e-12f);
  sh0 *= inv; sh1 *= inv;

  float prj = 0.f;
  for (int r = 0; r < kRFF; ++r) {
    float p = fmaf(omega[r * kShell + 2 * lane], sh0,
                   omega[r * kShell + 2 * lane + 1] * sh1);
    p = wave_sum32(p);
    if (lane == r) prj = p + rbias[r];
  }
  const float rscl = sqrtf(2.0f / (2.0f * kRFF)) * sqrtf(expf(-0.5f));
  const float cv = cosf(prj) * rscl;
  const float sv = sinf(prj) * rscl;

  const float mg[5] = {1.0f, 1.0f, 0.70710678f, 0.40824829f, 0.20412415f};
  float pc[5]; float qp = 1.0f;
#pragma unroll
  for (int m = 0; m < 5; ++m) { pc[m] = mg[m] * qp; qp *= q; }

  float ph0 = 0.f, ph1 = 0.f;
  for (int o = 0; o < 64; ++o) {
    const float* wr = Wcomb + (long long)o * 320;
    float a = 0.f;
#pragma unroll
    for (int m = 0; m < 5; ++m)
      a = fmaf(pc[m], fmaf(wr[m * 64 + lane], cv, wr[m * 64 + 32 + lane] * sv), a);
    a = wave_sum32(a);
    if (o < 32) { if (lane == o)      ph0 = a; }
    else        { if (lane == o - 32) ph1 = a; }
  }
  __bf16* po = phiOut + (long long)tok * 64;
  po[lane]      = to_bf16(ph0);
  po[lane + 32] = to_bf16(ph1);
}

// ---------------------------------------------------------------------------
// logits = (phi_tile @ phi^T) * Toeplitz * invTemp, causal softmax -> attn bf16
// grid: (N/16, B), block 256 (8 waves x 64 cols each)
// ---------------------------------------------------------------------------
__global__ __launch_bounds__(256) void logits_softmax_kernel(
    const __bf16* __restrict__ phi, __bf16* __restrict__ attn,
    const float* __restrict__ logTemp) {
  const int rowTile = blockIdx.x;
  const int batch   = blockIdx.y;
  const int wave = threadIdx.x >> 5;
  const int lane = threadIdx.x & 31;
  const int half = lane >> 4;
  const int nin  = lane & 15;

  const __bf16* phiB = phi + (long long)batch * kN * 64;
  const __bf16* At   = phiB + (long long)rowTile * 16 * 64;

  const float lt   = logTemp[0];
  const float invT = 1.0f / (log1pf(expf(lt)) + 0.1f);

  v8f z = {0.f,0.f,0.f,0.f,0.f,0.f,0.f,0.f};
  v8f acc[4] = {z, z, z, z};
#pragma unroll
  for (int k = 0; k < 64; k += 32) {
    v16bf a = load_a_frag(At, 64, k);
#pragma unroll
    for (int t = 0; t < 4; ++t) {
      const int nb = wave * 64 + t * 16;
      v16bf b = load_b_frag_nt(phiB + (long long)nb * 64, 64, k);
      acc[t] = wmma_bf16(a, b, acc[t]);
    }
  }

  const int rowBase = rowTile * 16;
  float sv[4][8];
  float pmax[8];
#pragma unroll
  for (int r = 0; r < 8; ++r) pmax[r] = -1e30f;
#pragma unroll
  for (int t = 0; t < 4; ++t) {
    const int col = wave * 64 + t * 16 + nin;
#pragma unroll
    for (int r = 0; r < 8; ++r) {
      const int row = rowBase + r + half * 8;
      float v;
      if (col <= row) {
        const float w = __expf(-0.03f * (float)(row - col));
        v = acc[t][r] * w * invT;
      } else {
        v = -1e30f;
      }
      sv[t][r] = v;
      pmax[r]  = fmaxf(pmax[r], v);
    }
  }
#pragma unroll
  for (int r = 0; r < 8; ++r) pmax[r] = half_max16(pmax[r]);

  __shared__ float red[8][16];
  __shared__ float rowStat[16];
  if (nin == 0) {
#pragma unroll
    for (int r = 0; r < 8; ++r) red[wave][r + half * 8] = pmax[r];
  }
  __syncthreads();
  if (threadIdx.x < 16) {
    float m = red[0][threadIdx.x];
#pragma unroll
    for (int w = 1; w < 8; ++w) m = fmaxf(m, red[w][threadIdx.x]);
    rowStat[threadIdx.x] = m;
  }
  __syncthreads();
  float rmax[8];
#pragma unroll
  for (int r = 0; r < 8; ++r) rmax[r] = rowStat[r + half * 8];

  float psum[8];
#pragma unroll
  for (int r = 0; r < 8; ++r) psum[r] = 0.f;
#pragma unroll
  for (int t = 0; t < 4; ++t)
#pragma unroll
    for (int r = 0; r < 8; ++r) {
      const float e = __expf(sv[t][r] - rmax[r]);
      sv[t][r] = e;
      psum[r] += e;
    }
#pragma unroll
  for (int r = 0; r < 8; ++r) psum[r] = half_sum16(psum[r]);
  __syncthreads();
  if (nin == 0) {
#pragma unroll
    for (int r = 0; r < 8; ++r) red[wave][r + half * 8] = psum[r];
  }
  __syncthreads();
  if (threadIdx.x < 16) {
    float s = 0.f;
#pragma unroll
    for (int w = 0; w < 8; ++w) s += red[w][threadIdx.x];
    rowStat[threadIdx.x] = 1.0f / s;
  }
  __syncthreads();
#pragma unroll
  for (int r = 0; r < 8; ++r) {
    const float rinv = rowStat[r + half * 8];
    const int   row  = rowBase + r + half * 8;
#pragma unroll
    for (int t = 0; t < 4; ++t) {
      const int col = wave * 64 + t * 16 + nin;
      attn[((long long)batch * kN + row) * kN + col] = to_bf16(sv[t][r] * rinv);
    }
  }
}

// ---------------------------------------------------------------------------
// Batched GEMM:  C[M,N] = A[M,K] * Bw[N,K]^T (+ bias[N]); bf16 in, f32 acc.
// grid: (M/256, N/64, batches); block 256; each wave owns a 32x64 tile.
// Shared 64x32 B strip is staged to LDS per K-step via the Tensor Data Mover
// (double-buffered, TENSORcnt-tracked), falling back to cooperative copies.
// ---------------------------------------------------------------------------
__global__ __launch_bounds__(256) void gemm_nt_bf16_kernel(
    int M, int N, int K,
    const __bf16* __restrict__ A, long long sAb,
    const __bf16* __restrict__ Bw, long long sBb,
    const float* __restrict__ bias,
    float* __restrict__ Cf, long long sCf,
    __bf16* __restrict__ Cb, long long sCb,
    int transposeCb) {
  __shared__ __align__(32) __bf16 Bsh[2][64 * 32];

  const int wave = threadIdx.x >> 5;
  const int lane = threadIdx.x & 31;
  const int half = lane >> 4;
  const int nin  = lane & 15;
  const int rowBase = (blockIdx.x * 8 + wave) * 32;   // 32 rows per wave
  const int nBase   = blockIdx.y * 64;
  const int batch   = blockIdx.z;

  const __bf16* Ab = A + (long long)batch * sAb + (long long)rowBase * K;
  const __bf16* Bt = Bw + (long long)batch * sBb + (long long)nBase * K;

  v8f z = {0.f,0.f,0.f,0.f,0.f,0.f,0.f,0.f};
  v8f acc[2][4] = {{z, z, z, z}, {z, z, z, z}};

  const int KT = K >> 5;
#if ASA_USE_TDM
  if (wave == 0)
    tdm_load_b_tile(Bt, (unsigned)(unsigned long long)&Bsh[0][0], K);
#else
  coop_load_b_tile(Bt, &Bsh[0][0], K);
#endif

  for (int kt = 0; kt < KT; ++kt) {
    const int cur = kt & 1;
#if ASA_USE_TDM
    if (wave == 0) __builtin_amdgcn_s_wait_tensorcnt(0);
#endif
    __syncthreads();   // tile `cur` visible; prior readers of `cur^1` are done
    if (kt + 1 < KT) {
#if ASA_USE_TDM
      if (wave == 0)
        tdm_load_b_tile(Bt + (kt + 1) * 32,
                        (unsigned)(unsigned long long)&Bsh[cur ^ 1][0], K);
#else
      coop_load_b_tile(Bt + (kt + 1) * 32, &Bsh[cur ^ 1][0], K);
#endif
    }
    v16bf a0 = load_a_frag(Ab, K, kt * 32);
    v16bf a1 = load_a_frag(Ab + (long long)16 * K, K, kt * 32);
#pragma unroll
    for (int t = 0; t < 4; ++t) {
      v16bf b = load_b_frag_lds(&Bsh[cur][0], t);
      acc[0][t] = wmma_bf16(a0, b, acc[0][t]);
      acc[1][t] = wmma_bf16(a1, b, acc[1][t]);
    }
    if (kt + 1 < KT)   // global_prefetch_b8 for the next A tile
      __builtin_prefetch((const void*)(Ab + (long long)nin * K + kt * 32 + 32), 0, 1);
  }

#pragma unroll
  for (int mi = 0; mi < 2; ++mi) {
#pragma unroll
    for (int t = 0; t < 4; ++t) {
      const int nG = nBase + t * 16 + nin;
      const float bb = bias ? bias[nG] : 0.0f;
#pragma unroll
      for (int r = 0; r < 8; ++r) {
        const int mG = rowBase + mi * 16 + r + half * 8;
        const float v = acc[mi][t][r] + bb;
        if (Cf) Cf[(long long)batch * sCf + (long long)mG * N + nG] = v;
        if (Cb) {
          const long long idx = transposeCb
              ? ((long long)batch * sCb + (long long)nG * M + mG)
              : ((long long)batch * sCb + (long long)mG * N + nG);
          Cb[idx] = to_bf16(v);
        }
      }
    }
  }
}

// ---------------------------------------------------------------------------
// Host side
// ---------------------------------------------------------------------------
extern "C" void kernel_launch(void* const* d_in, const int* in_sizes, int n_in,
                              void* d_out, int out_size, void* d_ws, size_t ws_size,
                              hipStream_t stream) {
  (void)in_sizes; (void)n_in; (void)out_size; (void)ws_size;
  const float* x        = (const float*)d_in[0];
  const float* W_charge = (const float*)d_in[1];
  const float* b_charge = (const float*)d_in[2];
  const float* W_shell  = (const float*)d_in[3];
  const float* b_shell  = (const float*)d_in[4];
  const float* omega    = (const float*)d_in[5];
  const float* rff_bias = (const float*)d_in[6];
  const float* W_comb   = (const float*)d_in[7];
  const float* W_v      = (const float*)d_in[8];
  const float* b_v      = (const float*)d_in[9];
  const float* W_out    = (const float*)d_in[10];
  const float* b_out    = (const float*)d_in[11];
  const float* log_temp = (const float*)d_in[12];
  float* out = (float*)d_out;

  const long long BN = (long long)kB * kN;   // 32768
  const long long nX = BN * kD;              // 33.55M
  const long long nW = (long long)kD * kD;   // 1.05M

  char* ws = (char*)d_ws;
  size_t off = 0;
  auto carve = [&](size_t bytes) -> void* {
    void* p = ws + off;
    off += (bytes + 255) & ~(size_t)255;
    return p;
  };
  __bf16* xb    = (__bf16*)carve(nX * 2);                      // x in bf16
  __bf16* Wvb   = (__bf16*)carve(nW * 2);
  __bf16* Woutb = (__bf16*)carve(nW * 2);
  __bf16* phib  = (__bf16*)carve(BN * 64 * 2);                 // phi (BN,64)
  __bf16* attnb = (__bf16*)carve((long long)kB * kN * kN * 2); // attn
  __bf16* Vtb   = (__bf16*)carve(nX * 2);                      // V^T (D,N)/batch
  __bf16* out1b = (__bf16*)carve(nX * 2);                      // attn@V (BN,D)

  cvt_f32_bf16_kernel<<<dim3((unsigned)((nX + 255) / 256)), 256, 0, stream>>>(x, xb, nX);
  cvt_f32_bf16_kernel<<<dim3((unsigned)((nW + 255) / 256)), 256, 0, stream>>>(W_v, Wvb, nW);
  cvt_f32_bf16_kernel<<<dim3((unsigned)((nW + 255) / 256)), 256, 0, stream>>>(W_out, Woutb, nW);

  features_kernel<<<dim3((unsigned)(BN / 8)), 256, 0, stream>>>(
      x, W_charge, b_charge, W_shell, b_shell, omega, rff_bias, W_comb, phib);

  // V = x @ W_v^T + b_v   (stored transposed bf16: per batch (D,N))
  gemm_nt_bf16_kernel<<<dim3(kN / 256, kD / 64, kB), 256, 0, stream>>>(
      kN, kD, kD,
      xb,  (long long)kN * kD,
      Wvb, 0,
      b_v,
      nullptr, 0,
      Vtb, (long long)kD * kN, /*transpose=*/1);

  logits_softmax_kernel<<<dim3(kN / 16, kB), 256, 0, stream>>>(phib, attnb, log_temp);

  // out1 = attn @ V        (B operand is V^T as (N=D, K=N))
  gemm_nt_bf16_kernel<<<dim3(kN / 256, kD / 64, kB), 256, 0, stream>>>(
      kN, kD, kN,
      attnb, (long long)kN * kN,
      Vtb,   (long long)kD * kN,
      nullptr,
      nullptr, 0,
      out1b, (long long)kN * kD, /*transpose=*/0);

  // out = out1 @ W_out^T + b_out  -> f32 d_out
  gemm_nt_bf16_kernel<<<dim3(kN / 256, kD / 64, kB), 256, 0, stream>>>(
      kN, kD, kD,
      out1b, (long long)kN * kD,
      Woutb, 0,
      b_out,
      out, (long long)kN * kD,
      nullptr, 0, /*transpose=*/0);
}
